// PointENAblock_73512660238918
// MI455X (gfx1250) — matrix operand order repaired and checked by reading, not compiled, not run on hardware
//
#include <hip/hip_runtime.h>
#include <math.h>

// Problem constants (match reference)
#define CDIM   128
#define KNN_K  8
#define BATCH  2
#define NPTS   8192
#define MROWS  (BATCH * NPTS)   // 16384

typedef __attribute__((ext_vector_type(16))) _Float16 v16h;
typedef __attribute__((ext_vector_type(8)))  _Float16 v8h;
typedef __attribute__((ext_vector_type(8)))  float    v8f;

// ---------------------------------------------------------------------------
// Kernel 1: convert feats and the 4 weight matrices to f16 (one-shot, tiny).
// ---------------------------------------------------------------------------
__global__ void cvt_f16_kernel(const float* __restrict__ feats,
                               const float* __restrict__ Wq,
                               const float* __restrict__ Wk,
                               const float* __restrict__ Wv,
                               const float* __restrict__ Wo,
                               _Float16* __restrict__ feats16,
                               _Float16* __restrict__ w16,
                               int nFeats) {
  int i = blockIdx.x * blockDim.x + threadIdx.x;
  int total = nFeats + 4 * CDIM * CDIM;
  if (i >= total) return;
  if (i < nFeats) {
    feats16[i] = (_Float16)feats[i];
  } else {
    int r = i - nFeats;
    int which = r / (CDIM * CDIM);
    int off   = r % (CDIM * CDIM);
    const float* src = (which == 0) ? Wq : (which == 1) ? Wk : (which == 2) ? Wv : Wo;
    w16[r] = (_Float16)src[off];
  }
}

// ---------------------------------------------------------------------------
// WMMA GEMM:  Y[m][n] = sum_k A[m][k] * W[n][k]   (i.e. y = x @ W^T)
// One 16x16 output tile per wave32; K marched in steps of 32 via
// v_wmma_f32_16x16x32_f16 with f32 accumulation.
// Fragment layouts per CDNA5 ISA 7.12.2.
// ---------------------------------------------------------------------------
__device__ __forceinline__ v16h load_A_frag(const _Float16* __restrict__ A,
                                            int M0, int k0, int lane) {
  int lp = lane & 15, hf = lane >> 4;
  const _Float16* row = A + (size_t)(M0 + lp) * CDIM;
  v8h lo = *(const v8h*)(row + k0 + 8 * hf);        // K = k0 + 8h .. +7
  v8h hi = *(const v8h*)(row + k0 + 16 + 8 * hf);   // K = k0+16+8h .. +7
  v16h r;
#pragma unroll
  for (int i = 0; i < 8; ++i) { r[i] = lo[i]; r[i + 8] = hi[i]; }
  return r;
}

__device__ __forceinline__ v16h load_B_frag(const _Float16* __restrict__ W,
                                            int N0, int k0, int lane) {
  int lp = lane & 15, hf = lane >> 4;
  // B[k][n] = W[N0+n][k0+k]; element j holds K = 8h + j (j<8), 16 + 8h + (j-8) (j>=8)
  const _Float16* row = W + (size_t)(N0 + lp) * CDIM;
  v8h lo = *(const v8h*)(row + k0 + 8 * hf);
  v8h hi = *(const v8h*)(row + k0 + 16 + 8 * hf);
  v16h r;
#pragma unroll
  for (int i = 0; i < 8; ++i) { r[i] = lo[i]; r[i + 8] = hi[i]; }
  return r;
}

__global__ void gemm_xWT_kernel(const _Float16* __restrict__ A,
                                const _Float16* __restrict__ W,
                                float* __restrict__ Y, int Mrows) {
  int wave = threadIdx.x >> 5;
  int lane = threadIdx.x & 31;
  int tile = blockIdx.x * (blockDim.x >> 5) + wave;
  const int ntn = CDIM / 16;                 // 8 tiles along N
  int total = (Mrows / 16) * ntn;
  if (tile >= total) return;
  int M0 = (tile / ntn) * 16;
  int N0 = (tile % ntn) * 16;

  v8f acc = {};
#pragma unroll
  for (int k0 = 0; k0 < CDIM; k0 += 32) {
    v16h a = load_A_frag(A, M0, k0, lane);
    v16h b = load_B_frag(W, N0, k0, lane);
    acc = __builtin_amdgcn_wmma_f32_16x16x32_f16(
        /*neg_a=*/false, a, /*neg_b=*/false, b,
        /*c_mod=*/(short)0, acc, /*reuse_a=*/false, /*reuse_b=*/false);
  }

  int lp = lane & 15, hf = lane >> 4;
#pragma unroll
  for (int r = 0; r < 8; ++r)
    Y[(size_t)(M0 + r + 8 * hf) * CDIM + N0 + lp] = acc[r];
}

// ---------------------------------------------------------------------------
// Kernel 3: exact KNN (K=8, includes self). One thread per query point.
// Candidate coords staged through LDS in tiles of 256; per-thread top-8 kept
// in registers with a branchless unrolled stable insertion (ties -> lower idx
// first, matching top_k on the streamed candidate order).
// ---------------------------------------------------------------------------
__global__ void knn_kernel(const float* __restrict__ coords,
                           int* __restrict__ idx_out,
                           float* __restrict__ dist_out) {
  __shared__ float sx[256], sy[256], sz[256];

  int q = blockIdx.x * blockDim.x + threadIdx.x;   // flattened (b*N + n)
  int b = q / NPTS;
  const float* cb = coords + (size_t)b * NPTS * 3;

  float qx = coords[(size_t)q * 3 + 0];
  float qy = coords[(size_t)q * 3 + 1];
  float qz = coords[(size_t)q * 3 + 2];

  float bd[KNN_K];
  int   bi[KNN_K];
#pragma unroll
  for (int j = 0; j < KNN_K; ++j) { bd[j] = 3.0e38f; bi[j] = 0; }

  for (int base = 0; base < NPTS; base += 256) {
    __syncthreads();
    int c = base + threadIdx.x;
    sx[threadIdx.x] = cb[(size_t)c * 3 + 0];
    sy[threadIdx.x] = cb[(size_t)c * 3 + 1];
    sz[threadIdx.x] = cb[(size_t)c * 3 + 2];
    __syncthreads();

    for (int t = 0; t < 256; ++t) {
      float dx = qx - sx[t];
      float dy = qy - sy[t];
      float dz = qz - sz[t];
      float d2 = dx * dx + dy * dy + dz * dz;
      if (d2 < bd[KNN_K - 1]) {          // rare path
        float dc = d2; int ic = base + t;
#pragma unroll
        for (int j = 0; j < KNN_K; ++j) { // stable bubble insert, all-register
          bool less = dc < bd[j];
          float dj = bd[j]; int ij = bi[j];
          if (less) { bd[j] = dc; bi[j] = ic; dc = dj; ic = ij; }
        }
      }
    }
  }

#pragma unroll
  for (int j = 0; j < KNN_K; ++j) {
    idx_out[(size_t)q * KNN_K + j]  = bi[j];
    dist_out[(size_t)q * KNN_K + j] = sqrtf(bd[j]);
  }
}

// ---------------------------------------------------------------------------
// Kernel 4: gated neighborhood attention. One wave32 per point; each lane owns
// 4 channels (C=128). Dots reduced with a full shuffle butterfly; softmax in
// fp32 (numerically stable); aggregated v written as f16 for the Wo GEMM.
// ---------------------------------------------------------------------------
__global__ void attn_kernel(const float* __restrict__ q32,
                            const float* __restrict__ k32,
                            const float* __restrict__ v32,
                            const int* __restrict__ nidx,
                            const float* __restrict__ ndist,
                            const float* __restrict__ log_gamma,
                            _Float16* __restrict__ agg16) {
  int wave = threadIdx.x >> 5;
  int lane = threadIdx.x & 31;
  int p = blockIdx.x * (blockDim.x >> 5) + wave;
  if (p >= MROWS) return;
  int b = p / NPTS;
  size_t browbase = (size_t)b * NPTS;

  float lg = log_gamma[0];
  const float scale = 0.0883883476483184f;   // 1/sqrt(128)

  float4 qv = *(const float4*)(q32 + (size_t)p * CDIM + lane * 4);
  qv.x *= scale; qv.y *= scale; qv.z *= scale; qv.w *= scale;

  float sc[KNN_K];
#pragma unroll
  for (int j = 0; j < KNN_K; ++j) {
    int nb = nidx[(size_t)p * KNN_K + j];
    float4 kv = *(const float4*)(k32 + (browbase + nb) * CDIM + lane * 4);
    float d = qv.x * kv.x + qv.y * kv.y + qv.z * kv.z + qv.w * kv.w;
#pragma unroll
    for (int off = 16; off >= 1; off >>= 1)
      d += __shfl_xor(d, off, 32);
    sc[j] = d * __expf(lg * ndist[(size_t)p * KNN_K + j]);
  }

  float mx = sc[0];
#pragma unroll
  for (int j = 1; j < KNN_K; ++j) mx = fmaxf(mx, sc[j]);
  float w[KNN_K]; float sum = 0.f;
#pragma unroll
  for (int j = 0; j < KNN_K; ++j) { w[j] = __expf(sc[j] - mx); sum += w[j]; }
  float inv = 1.f / sum;

  float a0 = 0.f, a1 = 0.f, a2 = 0.f, a3 = 0.f;
#pragma unroll
  for (int j = 0; j < KNN_K; ++j) {
    int nb = nidx[(size_t)p * KNN_K + j];
    float4 vv = *(const float4*)(v32 + (browbase + nb) * CDIM + lane * 4);
    float wj = w[j] * inv;
    a0 += wj * vv.x; a1 += wj * vv.y; a2 += wj * vv.z; a3 += wj * vv.w;
  }

  _Float16* out = agg16 + (size_t)p * CDIM + lane * 4;
  out[0] = (_Float16)a0; out[1] = (_Float16)a1;
  out[2] = (_Float16)a2; out[3] = (_Float16)a3;
}

// ---------------------------------------------------------------------------
// Host-side orchestration
// ---------------------------------------------------------------------------
extern "C" void kernel_launch(void* const* d_in, const int* in_sizes, int n_in,
                              void* d_out, int out_size, void* d_ws, size_t ws_size,
                              hipStream_t stream) {
  const float* feats  = (const float*)d_in[0];
  const float* coords = (const float*)d_in[1];
  const float* Wq     = (const float*)d_in[2];
  const float* Wk     = (const float*)d_in[3];
  const float* Wv     = (const float*)d_in[4];
  const float* Wo     = (const float*)d_in[5];
  const float* lgam   = (const float*)d_in[6];
  float* out = (float*)d_out;

  // Carve scratch (all 256B aligned)
  char* base = (char*)d_ws;
  size_t off = 0;
  auto carve = [&](size_t bytes) -> char* {
    off = (off + 255) & ~(size_t)255;
    char* p = base + off;
    off += bytes;
    return p;
  };
  _Float16* feats16 = (_Float16*)carve((size_t)MROWS * CDIM * 2);
  _Float16* w16     = (_Float16*)carve((size_t)4 * CDIM * CDIM * 2);
  float*    q32     = (float*)   carve((size_t)MROWS * CDIM * 4);
  float*    k32     = (float*)   carve((size_t)MROWS * CDIM * 4);
  float*    v32     = (float*)   carve((size_t)MROWS * CDIM * 4);
  int*      nidx    = (int*)     carve((size_t)MROWS * KNN_K * 4);
  float*    ndist   = (float*)   carve((size_t)MROWS * KNN_K * 4);
  _Float16* agg16   = (_Float16*)carve((size_t)MROWS * CDIM * 2);
  (void)ws_size;

  const int nFeats = MROWS * CDIM;

  // 1) f32 -> f16 conversion of activations + weights
  {
    int total = nFeats + 4 * CDIM * CDIM;
    int blocks = (total + 255) / 256;
    cvt_f16_kernel<<<blocks, 256, 0, stream>>>(feats, Wq, Wk, Wv, Wo,
                                               feats16, w16, nFeats);
  }

  // 2) q, k, v projections via WMMA (8192 wave-tiles each -> 1024 blocks)
  {
    int tiles  = (MROWS / 16) * (CDIM / 16);
    int blocks = (tiles + 7) / 8;
    gemm_xWT_kernel<<<blocks, 256, 0, stream>>>(feats16, w16 + 0 * CDIM * CDIM, q32, MROWS);
    gemm_xWT_kernel<<<blocks, 256, 0, stream>>>(feats16, w16 + 1 * CDIM * CDIM, k32, MROWS);
    gemm_xWT_kernel<<<blocks, 256, 0, stream>>>(feats16, w16 + 2 * CDIM * CDIM, v32, MROWS);
  }

  // 3) exact KNN (one thread per query point)
  knn_kernel<<<MROWS / 256, 256, 0, stream>>>(coords, nidx, ndist);

  // 4) gated neighborhood attention (one wave32 per point)
  attn_kernel<<<(MROWS + 7) / 8, 256, 0, stream>>>(q32, k32, v32, nidx, ndist,
                                                   lgam, agg16);

  // 5) output projection via WMMA, f32 result straight into d_out
  {
    int tiles  = (MROWS / 16) * (CDIM / 16);
    int blocks = (tiles + 7) / 8;
    gemm_xWT_kernel<<<blocks, 256, 0, stream>>>(agg16, w16 + 3 * CDIM * CDIM, out, MROWS);
  }
  (void)in_sizes; (void)n_in; (void)out_size;
}
